// SED_Attention_LSTM_44848048504856
// MI455X (gfx1250) — compile-verified, matrix operand order checked by reading, not run on hardware
//
#include <hip/hip_runtime.h>
#include <hip/hip_bf16.h>
#include <stdint.h>

typedef __bf16 bf16_t;
typedef __attribute__((ext_vector_type(16))) __bf16 v16bf;
typedef __attribute__((ext_vector_type(8)))  __bf16 v8bf;
typedef __attribute__((ext_vector_type(8)))  float  v8f;

#define B_   8
#define T_   64
#define MEL_ 128
#define L_   256
#define H_   256
#define D_   256
#define C_   10
#define HD_  65536   // H*D
#define FL_  1024    // 4*L

__device__ __forceinline__ float sigmf(float x) { return 1.0f / (1.0f + __expf(-x)); }

// Lane operand load for WMMA bf16 16x16x32: lane<16 holds K=0..7 / K=16..23,
// lane>=16 holds K=8..15 / K=24..31.  base = row start + k0.
__device__ __forceinline__ v16bf ld_bf16_row(const bf16_t* base, int kh) {
  union { v16bf v; v8bf h[2]; } u;
  u.h[0] = *(const v8bf*)(base + kh * 8);
  u.h[1] = *(const v8bf*)(base + 16 + kh * 8);
  return u.v;
}

// ---------------- generic bf16 WMMA GEMM ----------------
// C[M,N] = act(A[M,K] * B[N,K]^T + bias[N]);  M=M16*16 (M16 % MT == 0),
// N=N64*64, K%32==0.  One wave computes a (MT*16)x64 tile: B fragments are
// reused MT times (MT=2 => 1.5 loads/WMMA instead of 2.5).
template <int MT>
__global__ void gemm_bf16_kernel(const bf16_t* __restrict__ A, int lda,
                                 const bf16_t* __restrict__ B, int ldb,
                                 const float* __restrict__ bias,
                                 float* __restrict__ Cf, bf16_t* __restrict__ Cb,
                                 int ldc, int M16, int N64, int K, int act) {
  int w    = (blockIdx.x * blockDim.x + threadIdx.x) >> 5;
  int lane = threadIdx.x & 31;
  int mg   = M16 / MT;                   // m-groups
  if (w >= mg * N64) return;             // wave-uniform exit (EXEC all-1 for WMMA)
  int mt = (w % mg) * MT;                // consecutive waves share the B tile (L2 reuse)
  int nt = w / mg;
  int r  = lane & 15;
  int kh = lane >> 4;

  const bf16_t* arow[MT];
  #pragma unroll
  for (int m = 0; m < MT; ++m)
    arow[m] = A + (size_t)((mt + m) * 16 + r) * lda;
  const bf16_t* brow[4];
  #pragma unroll
  for (int j = 0; j < 4; ++j)
    brow[j] = B + (size_t)(nt * 64 + j * 16 + r) * ldb;

  v8f acc[MT][4] = {};
  for (int k0 = 0; k0 < K; k0 += 32) {
    v16bf a[MT], b[4];
    #pragma unroll
    for (int m = 0; m < MT; ++m) a[m] = ld_bf16_row(arow[m] + k0, kh);
    #pragma unroll
    for (int j = 0; j < 4; ++j)  b[j] = ld_bf16_row(brow[j] + k0, kh);
    #pragma unroll
    for (int m = 0; m < MT; ++m)
      #pragma unroll
      for (int j = 0; j < 4; ++j)
        acc[m][j] = __builtin_amdgcn_wmma_f32_16x16x32_bf16(
            false, a[m], false, b[j], (short)0, acc[m][j], false, false);
  }
  // C layout: lane&15 = column; element e = row e (+8 for lanes>=16)
  #pragma unroll
  for (int m = 0; m < MT; ++m) {
    int rowBase = (mt + m) * 16 + kh * 8;
    #pragma unroll
    for (int j = 0; j < 4; ++j) {
      int col = nt * 64 + j * 16 + (lane & 15);
      float bb = bias ? bias[col] : 0.0f;
      #pragma unroll
      for (int e = 0; e < 8; ++e) {
        float v = acc[m][j][e] + bb;
        if (act == 1) v = fmaxf(v, 0.0f);
        else if (act == 2) v = sigmf(v);
        size_t offc = (size_t)(rowBase + e) * ldc + col;
        if (Cf) Cf[offc] = v;
        if (Cb) Cb[offc] = (bf16_t)v;
      }
    }
  }
}

// ---------------- conversions / transpose ----------------
__global__ void f32_to_bf16_kernel(const float* __restrict__ s, bf16_t* __restrict__ d, int n) {
  int i = blockIdx.x * blockDim.x + threadIdx.x;
  if (i < n) d[i] = (bf16_t)s[i];
}

// xT[r=b*64+t, m] = x[b, m, t]  (bf16)
__global__ void transpose_x_kernel(const float* __restrict__ x, bf16_t* __restrict__ xT) {
  int i = blockIdx.x * blockDim.x + threadIdx.x;
  if (i >= B_ * T_ * MEL_) return;
  int m = i & (MEL_ - 1);
  int rr = i >> 7;
  int b = rr >> 6, t = rr & 63;
  xT[rr * MEL_ + m] = (bf16_t)x[((size_t)b * MEL_ + m) * T_ + t];
}

// ---------------- conv stack (fused conv+relu+pool) ----------------
__global__ void conv1_kernel(const float* __restrict__ f1, const float* __restrict__ w,
                             const float* __restrict__ bias, float* __restrict__ out) {
  int i = blockIdx.x * blockDim.x + threadIdx.x;
  if (i >= 512 * 16 * 512) return;
  int op = i & 511; int tmp = i >> 9; int ch = tmp & 15; int rr = tmp >> 4;
  const float* in = f1 + (size_t)rr * 1024;
  float w0 = w[ch * 3], w1 = w[ch * 3 + 1], w2 = w[ch * 3 + 2], bb = bias[ch];
  auto conv = [&](int p) {
    float s = bb + w1 * in[p];
    if (p > 0)    s += w0 * in[p - 1];
    if (p < 1023) s += w2 * in[p + 1];
    return fmaxf(s, 0.0f);
  };
  out[i] = fmaxf(conv(2 * op), conv(2 * op + 1));
}

__global__ void conv2_kernel(const float* __restrict__ p1, const float* __restrict__ w,
                             const float* __restrict__ bias, float* __restrict__ out) {
  int i = blockIdx.x * blockDim.x + threadIdx.x;
  if (i >= 512 * 32 * 256) return;
  int op = i & 255; int tmp = i >> 8; int ch = tmp & 31; int rr = tmp >> 5;
  auto conv = [&](int p) {
    float s = bias[ch];
    #pragma unroll
    for (int ci = 0; ci < 16; ++ci) {
      const float* in = p1 + ((size_t)rr * 16 + ci) * 512;
      const float* wb = w + (ch * 16 + ci) * 3;
      if (p > 0)   s += wb[0] * in[p - 1];
      s += wb[1] * in[p];
      if (p < 511) s += wb[2] * in[p + 1];
    }
    return fmaxf(s, 0.0f);
  };
  out[i] = fmaxf(conv(2 * op), conv(2 * op + 1));
}

// conv3: [512,32,256] -> feat_bf [512,256]  (no relu, no pool)
__global__ void conv3_kernel(const float* __restrict__ p2, const float* __restrict__ w,
                             const float* __restrict__ bias, bf16_t* __restrict__ feat_bf) {
  int i = blockIdx.x * blockDim.x + threadIdx.x;
  if (i >= 512 * 256) return;
  int p = i & 255; int rr = i >> 8;
  float s = bias[0];
  #pragma unroll
  for (int ci = 0; ci < 32; ++ci) {
    const float* in = p2 + ((size_t)rr * 32 + ci) * 256;
    const float* wb = w + ci * 3;
    if (p > 0)   s += wb[0] * in[p - 1];
    s += wb[1] * in[p];
    if (p < 255) s += wb[2] * in[p + 1];
  }
  feat_bf[i] = (bf16_t)s;
}

// ---------------- Wv/Wao fold (precompute) ----------------
// wfold[g,h] = sum_d Wao[d]*Wv[g*D+d, h];  bvw[g] = sum_d Wao[d]*bv[g*D+d]
__global__ void fold_wv_kernel(const float* __restrict__ Wv, const float* __restrict__ Wao,
                               const float* __restrict__ bv, float* __restrict__ wfold,
                               float* __restrict__ bvw) {
  int g = blockIdx.x, h = threadIdx.x;
  float acc = 0.0f;
  for (int d = 0; d < D_; ++d)
    acc += Wao[d] * Wv[((size_t)(g * D_ + d)) * H_ + h];
  wfold[g * H_ + h] = acc;
  if (h == 0) {
    float a2 = 0.0f;
    for (int d = 0; d < D_; ++d) a2 += Wao[d] * bv[g * D_ + d];
    bvw[g] = a2;
  }
}

// ---------------- recurrent-state kernels ----------------
__global__ void init_state_kernel(float* c, bf16_t* c_bf, bf16_t* h_bf) {
  int i = blockIdx.x * blockDim.x + threadIdx.x;
  if (i < 16 * H_) { c_bf[i] = (bf16_t)0.0f; h_bf[i] = (bf16_t)0.0f; }
  if (i < B_ * H_) c[i] = 0.0f;
}

// gates: gbuf holds (Wh*h + bh) rows b=0..7, cols 4H; xwx holds (Wx*f + bx).
__global__ void gate_kernel(const float* __restrict__ gbuf, const float* __restrict__ xwx,
                            float* __restrict__ c, bf16_t* __restrict__ c_bf,
                            float* __restrict__ og, int t) {
  int i = blockIdx.x * blockDim.x + threadIdx.x;
  if (i >= B_ * H_) return;
  int b = i >> 8, o = i & 255;
  const float* gb = gbuf + (size_t)b * FL_;
  const float* xr = xwx + ((size_t)(b * T_ + t)) * FL_;
  float gi = gb[o]          + xr[o];
  float gf = gb[H_ + o]     + xr[H_ + o];
  float gg = gb[2 * H_ + o] + xr[2 * H_ + o];
  float go = gb[3 * H_ + o] + xr[3 * H_ + o];
  float cn = sigmf(gf) * c[i] + sigmf(gi) * tanhf(gg);
  c[i] = cn;
  c_bf[b * H_ + o] = (bf16_t)cn;     // padded rows 8..15 stay zero from init
  og[i] = sigmf(go);
}

// vw[b,g] = og[b]·wfold[g] + bvw[g]
__global__ void vw_kernel(const float* __restrict__ og, const float* __restrict__ wfold,
                          const float* __restrict__ bvw, float* __restrict__ vw) {
  int b = blockIdx.x, g = threadIdx.x;
  float acc = bvw[g];
  const float* ob = og + b * H_;
  const float* wr = wfold + g * H_;
  for (int h = 0; h < H_; ++h) acc += ob[h] * wr[h];
  vw[b * H_ + g] = acc;
}

// fused scores(WMMA) -> softmax -> ·vw ; one wave per (b, 16-row h-tile).
// Softmax/dot split across both half-waves, combined via LDS.
__global__ void attn_kernel(const bf16_t* __restrict__ q_all, const bf16_t* __restrict__ k_bf,
                            const float* __restrict__ vw, const float* __restrict__ bao,
                            float* __restrict__ h_f, int t) {
  __shared__ float s[16][256];
  __shared__ float pm[2][16], pz[2][16], pa[2][16];
  int b = blockIdx.x >> 4;
  int ht = blockIdx.x & 15;
  int lane = threadIdx.x & 31;
  int r = lane & 15, kh = lane >> 4;
  const bf16_t* A  = q_all + ((size_t)(b * T_ + t)) * HD_;  // [H,D] row-major
  const bf16_t* Bk = k_bf + (size_t)b * HD_;                // [H,D] row-major
  const bf16_t* arow = A + (size_t)(ht * 16 + r) * D_;
  for (int nt = 0; nt < 16; ++nt) {
    v8f acc = {};
    const bf16_t* brow = Bk + (size_t)(nt * 16 + r) * D_;
    #pragma unroll
    for (int k0 = 0; k0 < D_; k0 += 32) {
      v16bf a  = ld_bf16_row(arow + k0, kh);
      v16bf bb = ld_bf16_row(brow + k0, kh);
      acc = __builtin_amdgcn_wmma_f32_16x16x32_bf16(false, a, false, bb, (short)0, acc, false, false);
    }
    #pragma unroll
    for (int e = 0; e < 8; ++e)
      s[kh * 8 + e][nt * 16 + (lane & 15)] = acc[e] * 0.0625f;  // 1/sqrt(256)
  }
  __syncthreads();
  {
    int row = lane & 15, half = lane >> 4;
    const float* srow = &s[row][half * 128];
    const float* vwb = vw + b * H_ + half * 128;
    float m = -1e30f;
    for (int g = 0; g < 128; ++g) m = fmaxf(m, srow[g]);
    float Z = 0.0f, accum = 0.0f;
    for (int g = 0; g < 128; ++g) {
      float e = __expf(srow[g] - m);
      Z += e; accum += e * vwb[g];
    }
    pm[half][row] = m; pz[half][row] = Z; pa[half][row] = accum;
  }
  __syncthreads();
  if (lane < 16) {
    int row = lane;
    float m0 = pm[0][row], m1 = pm[1][row];
    float m = fmaxf(m0, m1);
    float e0 = __expf(m0 - m), e1 = __expf(m1 - m);
    float Z = pz[0][row] * e0 + pz[1][row] * e1;
    float accum = pa[0][row] * e0 + pa[1][row] * e1;
    h_f[b * H_ + ht * 16 + row] = accum / Z + bao[0];
  }
}

// refresh padded h_bf and emit y[b,c] for this timestep
__global__ void hpost_kernel(const float* __restrict__ h_f, bf16_t* __restrict__ h_bf,
                             const float* __restrict__ Wout, const float* __restrict__ bout,
                             float* __restrict__ out, int t) {
  int i = blockIdx.x * blockDim.x + threadIdx.x;
  if (i < 16 * H_) {
    int row = i >> 8, col = i & 255;
    h_bf[i] = (row < B_) ? (bf16_t)h_f[row * H_ + col] : (bf16_t)0.0f;
  }
  if (i < B_ * C_) {
    int b = i / C_, cc = i % C_;
    float acc = bout[cc];
    const float* hr = h_f + b * H_;
    const float* wr = Wout + cc * H_;
    for (int h = 0; h < H_; ++h) acc += hr[h] * wr[h];
    out[(size_t)(b * C_ + cc) * T_ + t] = acc;
  }
}

// ---------------- host orchestration ----------------
extern "C" void kernel_launch(void* const* d_in, const int* in_sizes, int n_in,
                              void* d_out, int out_size, void* d_ws, size_t ws_size,
                              hipStream_t stream) {
  const float* x      = (const float*)d_in[0];
  const float* w_init = (const float*)d_in[1];
  const float* b_init = (const float*)d_in[2];
  const float* c1w = (const float*)d_in[3];   const float* c1b = (const float*)d_in[4];
  const float* c2w = (const float*)d_in[5];   const float* c2b = (const float*)d_in[6];
  const float* c3w = (const float*)d_in[7];   const float* c3b = (const float*)d_in[8];
  const float* Wx  = (const float*)d_in[9];   const float* bx  = (const float*)d_in[10];
  const float* Wh  = (const float*)d_in[11];  const float* bh  = (const float*)d_in[12];
  const float* Wxo = (const float*)d_in[13];  const float* bxo = (const float*)d_in[14];
  const float* Wq  = (const float*)d_in[15];  const float* bq  = (const float*)d_in[16];
  const float* Wk  = (const float*)d_in[17];  const float* bk  = (const float*)d_in[18];
  const float* Wv  = (const float*)d_in[19];  const float* bv  = (const float*)d_in[20];
  const float* Wao = (const float*)d_in[21];  const float* bao = (const float*)d_in[22];
  const float* Wout= (const float*)d_in[23];  const float* bout= (const float*)d_in[24];
  float* out = (float*)d_out;

  // workspace carve (256B aligned)
  size_t off = 0;
  auto alloc = [&](size_t bytes) -> void* {
    void* p = (char*)d_ws + off;
    off += (bytes + 255) & ~(size_t)255;
    return p;
  };
  bf16_t* wq_bf   = (bf16_t*)alloc((size_t)HD_ * H_ * 2);      // 32 MB
  bf16_t* wk_bf   = (bf16_t*)alloc((size_t)HD_ * H_ * 2);      // 32 MB
  bf16_t* q_all   = (bf16_t*)alloc((size_t)512 * HD_ * 2);     // 64 MB
  bf16_t* winit_bf= (bf16_t*)alloc((size_t)FL_ * MEL_ * 2);
  bf16_t* wx_bf   = (bf16_t*)alloc((size_t)FL_ * L_ * 2);
  bf16_t* wh_bf   = (bf16_t*)alloc((size_t)FL_ * H_ * 2);
  bf16_t* wxo_bf  = (bf16_t*)alloc((size_t)H_ * L_ * 2);
  bf16_t* xT_bf   = (bf16_t*)alloc((size_t)512 * MEL_ * 2);
  float*  f1      = (float*) alloc((size_t)512 * FL_ * 4);
  float*  p1      = (float*) alloc((size_t)512 * 16 * 512 * 4);
  float*  p2      = (float*) alloc((size_t)512 * 32 * 256 * 4);
  bf16_t* feat_bf = (bf16_t*)alloc((size_t)512 * L_ * 2);
  float*  xwx     = (float*) alloc((size_t)512 * FL_ * 4);
  bf16_t* xo_bf   = (bf16_t*)alloc((size_t)512 * H_ * 2);
  float*  wfold   = (float*) alloc((size_t)H_ * H_ * 4);
  float*  bvw     = (float*) alloc((size_t)H_ * 4);
  float*  cbuf    = (float*) alloc((size_t)B_ * H_ * 4);
  bf16_t* c_bf    = (bf16_t*)alloc((size_t)16 * H_ * 2);       // 16-row padded
  bf16_t* h_bf    = (bf16_t*)alloc((size_t)16 * H_ * 2);       // 16-row padded
  float*  og      = (float*) alloc((size_t)B_ * H_ * 4);
  float*  gbuf    = (float*) alloc((size_t)16 * FL_ * 4);
  bf16_t* k_bf    = (bf16_t*)alloc((size_t)16 * HD_ * 2);      // 16-row padded
  float*  vwbuf   = (float*) alloc((size_t)B_ * H_ * 4);
  float*  h_f     = (float*) alloc((size_t)B_ * H_ * 4);

  auto cvt = [&](const float* s, bf16_t* d, int n) {
    f32_to_bf16_kernel<<<(n + 255) / 256, 256, 0, stream>>>(s, d, n);
  };
  // --- precompute: weight conversions ---
  cvt(w_init, winit_bf, FL_ * MEL_);
  cvt(Wx, wx_bf, FL_ * L_);
  cvt(Wh, wh_bf, FL_ * H_);
  cvt(Wxo, wxo_bf, H_ * L_);
  cvt(Wq, wq_bf, HD_ * H_);
  cvt(Wk, wk_bf, HD_ * H_);

  auto gemm = [&](const bf16_t* A, int lda, const bf16_t* B, int ldb, const float* bias,
                  float* Cf, bf16_t* Cb, int ldc, int M16, int N64, int K, int act, int MT) {
    int waves = (M16 / MT) * N64;
    int blocks = (waves + 7) / 8;
    if (MT == 2)
      gemm_bf16_kernel<2><<<blocks, 256, 0, stream>>>(A, lda, B, ldb, bias, Cf, Cb, ldc, M16, N64, K, act);
    else
      gemm_bf16_kernel<1><<<blocks, 256, 0, stream>>>(A, lda, B, ldb, bias, Cf, Cb, ldc, M16, N64, K, act);
  };

  // --- feature extractor ---
  transpose_x_kernel<<<(512 * MEL_ + 255) / 256, 256, 0, stream>>>(x, xT_bf);
  gemm(xT_bf, MEL_, winit_bf, MEL_, b_init, f1, nullptr, FL_, 32, 16, MEL_, 1, 2);   // relu
  conv1_kernel<<<(512 * 16 * 512 + 255) / 256, 256, 0, stream>>>(f1, c1w, c1b, p1);
  conv2_kernel<<<(512 * 32 * 256 + 255) / 256, 256, 0, stream>>>(p1, c2w, c2b, p2);
  conv3_kernel<<<(512 * 256 + 255) / 256, 256, 0, stream>>>(p2, c3w, c3b, feat_bf);

  // --- hoisted (non-recurrent) projections ---
  gemm(feat_bf, L_, wx_bf, L_, bx, xwx, nullptr, FL_, 32, 16, L_, 0, 2);             // Wx·f + bx
  gemm(feat_bf, L_, wxo_bf, L_, bxo, nullptr, xo_bf, H_, 32, 4, L_, 2, 2);           // sigmoid
  gemm(xo_bf, H_, wq_bf, H_, bq, nullptr, q_all, HD_, 32, 1024, H_, 0, 2);           // q for all t
  fold_wv_kernel<<<H_, H_, 0, stream>>>(Wv, Wao, bv, wfold, bvw);                    // kill Wv

  // --- recurrent scan ---
  init_state_kernel<<<(16 * H_ + 255) / 256, 256, 0, stream>>>(cbuf, c_bf, h_bf);
  for (int t = 0; t < T_; ++t) {
    gemm(h_bf, H_, wh_bf, H_, bh, gbuf, nullptr, FL_, 1, 16, H_, 0, 1);              // Wh·h + bh
    gate_kernel<<<(B_ * H_ + 255) / 256, 256, 0, stream>>>(gbuf, xwx, cbuf, c_bf, og, t);
    gemm(c_bf, H_, wk_bf, H_, bk, nullptr, k_bf, HD_, 1, 1024, H_, 0, 1);            // k = c·Wkᵀ
    vw_kernel<<<B_, H_, 0, stream>>>(og, wfold, bvw, vwbuf);
    attn_kernel<<<B_ * 16, 32, 0, stream>>>(q_all, k_bf, vwbuf, bao, h_f, t);
    hpost_kernel<<<(16 * H_ + 255) / 256, 256, 0, stream>>>(h_f, h_bf, Wout, bout, out, t);
  }
}
